// TuneableAttentionMHA_4063039062531
// MI455X (gfx1250) — matrix operand order repaired and checked by reading, not compiled, not run on hardware
//
#include <hip/hip_runtime.h>
#include <hip/hip_bf16.h>

// Problem constants (match reference)
#define B_  2
#define N_  2048
#define D_  1024
#define H_  16
#define R_  32
#define DK_ 64

typedef __attribute__((ext_vector_type(16))) _Float16 v16h;
typedef __attribute__((ext_vector_type(8)))  _Float16 v8h;
typedef __attribute__((ext_vector_type(8)))  float    v8f;
typedef __attribute__((ext_vector_type(4)))  int      v4i;

// ---------------------------------------------------------------------------
// gfx1250 async global->LDS path (ASYNCcnt), with a synchronous fallback so
// the file compiles on toolchains that don't declare the builtins.
// Builtin signature (from compiler diagnostic): both pointer params are
// int4* — global src in AS(1), LDS dst in AS(3).
// ---------------------------------------------------------------------------
#if defined(__AMDGCN__) &&                                                    \
    __has_builtin(__builtin_amdgcn_global_load_async_to_lds_b128) &&          \
    __has_builtin(__builtin_amdgcn_s_wait_asynccnt)
#define USE_ASYNC_LDS 1
#else
#define USE_ASYNC_LDS 0
#endif

// Copy 16 bytes (8 halfs) global -> LDS. Async variant issues
// GLOBAL_LOAD_ASYNC_TO_LDS_B128 (no VGPR data, tracked by ASYNCcnt).
__device__ __forceinline__ void async_copy_b128(const _Float16* g, _Float16* l) {
#if USE_ASYNC_LDS
  __builtin_amdgcn_global_load_async_to_lds_b128(
      (__attribute__((address_space(1))) v4i*)(g),
      (__attribute__((address_space(3))) v4i*)(l), 0, 0);
#else
  *(v8h*)l = *(const v8h*)g;
#endif
}

__device__ __forceinline__ void async_wait0() {
#if USE_ASYNC_LDS
  __builtin_amdgcn_s_wait_asynccnt(0);
#endif
}

__device__ __forceinline__ v8f wmma16(v16h a, v16h b, v8f c) {
  // D(f32 16x16) = A(f16 16x32) x B(f16 32x16) + C
  return __builtin_amdgcn_wmma_f32_16x16x32_f16(false, a, false, b, (short)0, c,
                                                false, false);
}

// A fragment (16x32 f16) from row-major [16 x ld] (halfs).
// Lane layout (ISA 7.12.2): lane<16 -> row=lane, K in {0..7, 16..23};
// lane>=16 -> row=lane-16, K in {8..15, 24..31}.
__device__ __forceinline__ v16h load_frag_a(const _Float16* base, int ld) {
  const int lane = threadIdx.x & 31;
  const int m  = lane & 15;
  const int kb = (lane >> 4) << 3;
  const _Float16* p = base + m * ld + kb;
  v8h lo = *(const v8h*)(p);
  v8h hi = *(const v8h*)(p + 16);
  v16h a;
#pragma unroll
  for (int i = 0; i < 8; ++i) { a[i] = lo[i]; a[i + 8] = hi[i]; }
  return a;
}

// B fragment (32x16 f16, KxN) from Bt stored row-major [16(N) x ld] halfs,
// Bt[n][k] == B[k][n]. Lane holds column n=lane&15, K in {kb..kb+7, kb+16..kb+23}, kb=(lane>>4)*8.
__device__ __forceinline__ v16h load_frag_b(const _Float16* bt, int ld) {
  const int lane = threadIdx.x & 31;
  const int n  = lane & 15;
  const int kb = (lane >> 4) << 3;
  const _Float16* p = bt + n * ld + kb;
  v8h lo = *(const v8h*)(p);
  v8h hi = *(const v8h*)(p + 16);
  v16h r;
#pragma unroll
  for (int i = 0; i < 8; ++i) { r[i] = lo[i]; r[i + 8] = hi[i]; }
  return r;
}

// C/D fragment (16x16 f32): vgpr i holds row i + 8*(lane>=16), col = lane&15.
__device__ __forceinline__ void store_frag_c_f16(_Float16* base, int ld, v8f c) {
  const int lane = threadIdx.x & 31;
  const int n  = lane & 15;
  const int r0 = (lane >> 4) << 3;
#pragma unroll
  for (int i = 0; i < 8; ++i) base[(r0 + i) * ld + n] = (_Float16)c[i];
}

__device__ __forceinline__ void store_frag_c_f32(float* base, int ld, v8f c) {
  const int lane = threadIdx.x & 31;
  const int n  = lane & 15;
  const int r0 = (lane >> 4) << 3;
#pragma unroll
  for (int i = 0; i < 8; ++i) base[(r0 + i) * ld + n] = c[i];
}

__global__ void cvt_f32_to_f16(const float* __restrict__ in,
                               _Float16* __restrict__ out, int n) {
  int i = blockIdx.x * blockDim.x + threadIdx.x;
  if (i < n) out[i] = (_Float16)in[i];
}

// ---------------------------------------------------------------------------
// Stage 1: fused QKV projection with double-buffered async LDS staging.
// grid = (B*N/128, H, 3). Block 256 = 8 waves; wave w owns token rows w*16..+15.
// Computes P[128,64] = X_tile @ W[head]^T with a K-loop of 32, pipelined:
//   wait(chunk i) -> barrier -> issue async(chunk i+1) -> 4 WMMAs on chunk i.
// z==0/1 (q/k): additionally multiply P[16,64] @ U[64,32] per wave (fused).
// z==2 (v):     store P as f16 into v[b,h,n,dk].
// ---------------------------------------------------------------------------
__global__ __launch_bounds__(256)
void qkv_kernel(const _Float16* __restrict__ x,
                const _Float16* __restrict__ Wq,
                const _Float16* __restrict__ Wk,
                const _Float16* __restrict__ Wv,
                const _Float16* __restrict__ U,
                _Float16* __restrict__ qo,
                _Float16* __restrict__ ko,
                _Float16* __restrict__ vo) {
  __shared__ __align__(32) _Float16 xs[2][128][32];  // X tile, double-buffer 16KB
  __shared__ __align__(32) _Float16 wsm[2][64][32];  // W rows (=Bt), dbl-buf   8KB
  __shared__ __align__(32) _Float16 us[32][64];      // U^T (r, dk)             4KB
  __shared__ __align__(32) _Float16 ps[8][16][64];   // per-wave P scratch     16KB

  const int tid   = threadIdx.x;
  const int w     = tid >> 5;
  const int h     = blockIdx.y;
  const int which = blockIdx.z;                      // 0=q 1=k 2=v (uniform)
  const _Float16* W = (which == 0) ? Wq : (which == 1) ? Wk : Wv;
  const int ch  = h * DK_;                           // this head's 64 cols of W
  const int gm0 = blockIdx.x * 128;                  // flat row b*N + n

  if (tid < DK_) {                                   // stage U^T once
#pragma unroll
    for (int r = 0; r < R_; ++r) us[r][tid] = U[tid * R_ + r];
  }

  const int row = tid >> 1, col = (tid & 1) << 4;    // staging assignment
  const _Float16* xrow = x + (size_t)(gm0 + row) * D_ + col;
  const _Float16* wrow = W + (size_t)(ch + row) * D_ + col;

  auto stage = [&](int buf, int k0) {
    async_copy_b128(xrow + k0,     &xs[buf][row][col]);
    async_copy_b128(xrow + k0 + 8, &xs[buf][row][col + 8]);
    if (tid < 128) {
      async_copy_b128(wrow + k0,     &wsm[buf][row][col]);
      async_copy_b128(wrow + k0 + 8, &wsm[buf][row][col + 8]);
    }
  };

  v8f acc[4] = {};
  stage(0, 0);                                       // prologue
  const int NCH = D_ / 32;
  for (int i = 0; i < NCH; ++i) {
    const int bb = i & 1;
    async_wait0();                                   // my async loads landed
    __syncthreads();                                 // everyone's landed
    if (i + 1 < NCH) stage(bb ^ 1, (i + 1) * 32);    // overlap next chunk
    v16h a = load_frag_a(&xs[bb][w * 16][0], 32);
#pragma unroll
    for (int nt = 0; nt < 4; ++nt) {
      v16h b = load_frag_b(&wsm[bb][nt * 16][0], 32);
      acc[nt] = wmma16(a, b, acc[nt]);
    }
    __syncthreads();                                 // done reading buf bb
  }

  const int gm = gm0 + w * 16;
  const int b  = gm / N_;
  const int n  = gm - b * N_;

  if (which == 2) {
    _Float16* dst = vo + (((size_t)b * H_ + h) * N_ + n) * DK_;
#pragma unroll
    for (int nt = 0; nt < 4; ++nt) store_frag_c_f16(dst + nt * 16, DK_, acc[nt]);
  } else {
    // fused low-rank projection: Q[16,32] = P[16,64] @ U[64,32]
    _Float16* myps = &ps[w][0][0];
#pragma unroll
    for (int nt = 0; nt < 4; ++nt) store_frag_c_f16(myps + nt * 16, 64, acc[nt]);
    __syncthreads();
    v16h a0 = load_frag_a(myps, 64);        // dk = 0..31
    v16h a1 = load_frag_a(myps + 32, 64);   // dk = 32..63
    _Float16* dst =
        ((which == 0) ? qo : ko) + (((size_t)b * H_ + h) * N_ + n) * R_;
#pragma unroll
    for (int rt = 0; rt < 2; ++rt) {
      v16h b0 = load_frag_b(&us[rt * 16][0], 64);
      v16h b1 = load_frag_b(&us[rt * 16][32], 64);
      v8f qa = {};
      qa = wmma16(a0, b0, qa);
      qa = wmma16(a1, b1, qa);
      store_frag_c_f16(dst + rt * 16, R_, qa);
    }
  }
}

// ---------------------------------------------------------------------------
// Stage 2: causal flash attention (ALiBi is identically zero under the causal
// mask since dist = max(j-i,0) == 0 for j<=i, so only causal masking remains).
// grid = (N/64, H, B). Block 128 = 4 waves; wave w owns q rows qm0+w*16..+15.
// K block staged via async b128 copies; V transposed into LDS (scatter).
// S = Q(16x32r) K^T via one WMMA per 16 keys; online softmax via half-wave
// shfl_xor reductions; O += P(16x32) V(32x64) via 4 WMMAs.
// ---------------------------------------------------------------------------
__global__ __launch_bounds__(128)
void attn_kernel(const _Float16* __restrict__ q,
                 const _Float16* __restrict__ k,
                 const _Float16* __restrict__ v,
                 _Float16* __restrict__ z) {
  __shared__ __align__(32) _Float16 ks[32][32];    // K block, row-major (=Bt)
  __shared__ __align__(32) _Float16 vst[64][32];   // V block transposed (=Bt)
  __shared__ __align__(32) _Float16 ps[4][16][32]; // per-wave P (A relayout)

  const int tid  = threadIdx.x;
  const int w    = tid >> 5;
  const int lane = tid & 31;
  const int grp  = lane >> 4;
  const int ln   = lane & 15;
  const int h    = blockIdx.y;
  const int b    = blockIdx.z;
  const int qm0  = blockIdx.x * 64;

  const _Float16* qh = q + (((size_t)b * H_ + h) * N_) * R_;
  const _Float16* kh = k + (((size_t)b * H_ + h) * N_) * R_;
  const _Float16* vh = v + (((size_t)b * H_ + h) * N_) * DK_;

  v16h aq = load_frag_a(qh + (size_t)(qm0 + w * 16) * R_, R_);

  v8f o[4] = {};
  float mrow[8], lrow[8];
#pragma unroll
  for (int e = 0; e < 8; ++e) { mrow[e] = -1e30f; lrow[e] = 0.0f; }

  const int qrow_base = qm0 + w * 16 + grp * 8;    // row of vgpr-elem e is +e
  const int nkb = (qm0 >> 5) + 2;                  // causal key-block bound

  for (int kb = 0; kb < nkb; ++kb) {
    __syncthreads();                               // prior consumers done
    {
      int row = tid >> 2, col = (tid & 3) << 3;    // 32x32 K block (async)
      async_copy_b128(kh + (size_t)(kb * 32 + row) * R_ + col, &ks[row][col]);
      int key = tid & 31, d0 = (tid >> 5) << 4;    // V transposed into LDS
      const _Float16* vrow = vh + (size_t)(kb * 32 + key) * DK_ + d0;
#pragma unroll
      for (int i = 0; i < 16; ++i) vst[d0 + i][key] = vrow[i];
    }
    async_wait0();
    __syncthreads();

    v8f s0 = {}, s1 = {};
    s0 = wmma16(aq, load_frag_b(&ks[0][0], 32), s0);
    s1 = wmma16(aq, load_frag_b(&ks[16][0], 32), s1);

    // scale + causal mask (element (e): row qrow_base+e, key kc0/kc1)
    const int kc0 = kb * 32 + ln;
    const int kc1 = kc0 + 16;
    float sv0[8], sv1[8];
#pragma unroll
    for (int e = 0; e < 8; ++e) {
      int qi = qrow_base + e;
      sv0[e] = (kc0 <= qi) ? s0[e] * 0.125f : -1e30f;
      sv1[e] = (kc1 <= qi) ? s1[e] * 0.125f : -1e30f;
    }
    // row-max across the 16 lanes of each half-wave group
    float rm[8];
#pragma unroll
    for (int e = 0; e < 8; ++e) rm[e] = fmaxf(sv0[e], sv1[e]);
#pragma unroll
    for (int off = 1; off < 16; off <<= 1)
#pragma unroll
      for (int e = 0; e < 8; ++e) rm[e] = fmaxf(rm[e], __shfl_xor(rm[e], off, 32));

    float p0[8], p1[8];
#pragma unroll
    for (int e = 0; e < 8; ++e) {
      float mn    = fmaxf(mrow[e], rm[e]);
      float alpha = __expf(mrow[e] - mn);
      mrow[e] = mn;
      lrow[e] *= alpha;
#pragma unroll
      for (int dt = 0; dt < 4; ++dt) o[dt][e] *= alpha;
      p0[e] = __expf(sv0[e] - mn);
      p1[e] = __expf(sv1[e] - mn);
    }
    float rs[8];
#pragma unroll
    for (int e = 0; e < 8; ++e) rs[e] = p0[e] + p1[e];
#pragma unroll
    for (int off = 1; off < 16; off <<= 1)
#pragma unroll
      for (int e = 0; e < 8; ++e) rs[e] += __shfl_xor(rs[e], off, 32);
#pragma unroll
    for (int e = 0; e < 8; ++e) lrow[e] += rs[e];

    // C-layout -> A-layout via per-wave LDS (LDS ops from a wave are in-order)
#pragma unroll
    for (int e = 0; e < 8; ++e) {
      ps[w][grp * 8 + e][ln]      = (_Float16)p0[e];
      ps[w][grp * 8 + e][16 + ln] = (_Float16)p1[e];
    }
    v16h ap = load_frag_a(&ps[w][0][0], 32);
#pragma unroll
    for (int dt = 0; dt < 4; ++dt) {
      v16h bv = load_frag_b(&vst[dt * 16][0], 32);
      o[dt] = wmma16(ap, bv, o[dt]);
    }
  }

  float inv[8];
#pragma unroll
  for (int e = 0; e < 8; ++e) inv[e] = 1.0f / lrow[e];
#pragma unroll
  for (int dt = 0; dt < 4; ++dt)
#pragma unroll
    for (int e = 0; e < 8; ++e) o[dt][e] *= inv[e];

  // z layout [B, N, H*DK] so the final projection is a plain GEMM
  _Float16* dst = z + ((size_t)b * N_ + qm0 + w * 16) * D_ + h * DK_;
#pragma unroll
  for (int dt = 0; dt < 4; ++dt) store_frag_c_f16(dst + dt * 16, D_, o[dt]);
}

// ---------------------------------------------------------------------------
// Stage 3: out = z @ Wproj^T, f32 epilogue straight to d_out.
// grid = (B*N/128, D/64). Block 256 = 8 waves. Same async double-buffering.
// ---------------------------------------------------------------------------
__global__ __launch_bounds__(256)
void proj_kernel(const _Float16* __restrict__ zin,
                 const _Float16* __restrict__ Wp,
                 float* __restrict__ out) {
  __shared__ __align__(32) _Float16 xs[2][128][32];
  __shared__ __align__(32) _Float16 wsm[2][64][32];
  const int tid = threadIdx.x;
  const int w   = tid >> 5;
  const int c0  = blockIdx.y * 64;
  const int gm0 = blockIdx.x * 128;

  const int row = tid >> 1, col = (tid & 1) << 4;
  const _Float16* zrow = zin + (size_t)(gm0 + row) * D_ + col;
  const _Float16* wrow = Wp + (size_t)(c0 + row) * D_ + col;

  auto stage = [&](int buf, int k0) {
    async_copy_b128(zrow + k0,     &xs[buf][row][col]);
    async_copy_b128(zrow + k0 + 8, &xs[buf][row][col + 8]);
    if (tid < 128) {
      async_copy_b128(wrow + k0,     &wsm[buf][row][col]);
      async_copy_b128(wrow + k0 + 8, &wsm[buf][row][col + 8]);
    }
  };

  v8f acc[4] = {};
  stage(0, 0);
  const int NCH = D_ / 32;
  for (int i = 0; i < NCH; ++i) {
    const int bb = i & 1;
    async_wait0();
    __syncthreads();
    if (i + 1 < NCH) stage(bb ^ 1, (i + 1) * 32);
    v16h a = load_frag_a(&xs[bb][w * 16][0], 32);
#pragma unroll
    for (int nt = 0; nt < 4; ++nt) {
      v16h b = load_frag_b(&wsm[bb][nt * 16][0], 32);
      acc[nt] = wmma16(a, b, acc[nt]);
    }
    __syncthreads();
  }
  float* dst = out + (size_t)(gm0 + w * 16) * D_ + c0;
#pragma unroll
  for (int nt = 0; nt < 4; ++nt) store_frag_c_f32(dst + nt * 16, D_, acc[nt]);
}

extern "C" void kernel_launch(void* const* d_in, const int* in_sizes, int n_in,
                              void* d_out, int out_size, void* d_ws, size_t ws_size,
                              hipStream_t stream) {
  // setup_inputs order: x, Wq, Wk, Wv, U, Wproj, mask, h, rel_bias_tokens
  const float* x  = (const float*)d_in[0];
  const float* Wq = (const float*)d_in[1];
  const float* Wk = (const float*)d_in[2];
  const float* Wv = (const float*)d_in[3];
  const float* U  = (const float*)d_in[4];
  const float* Wp = (const float*)d_in[5];
  float* out = (float*)d_out;

  char* p = (char*)d_ws;
  auto carve = [&](size_t nelem) {
    _Float16* r = (_Float16*)p;
    p += (nelem * sizeof(_Float16) + 255) & ~(size_t)255;
    return r;
  };
  const size_t BN = (size_t)B_ * N_;
  _Float16* x16  = carve(BN * D_);
  _Float16* Wq16 = carve((size_t)D_ * D_);
  _Float16* Wk16 = carve((size_t)D_ * D_);
  _Float16* Wv16 = carve((size_t)D_ * D_);
  _Float16* Wp16 = carve((size_t)D_ * D_);
  _Float16* U16  = carve((size_t)DK_ * R_);
  _Float16* q16  = carve((size_t)B_ * H_ * N_ * R_);
  _Float16* k16  = carve((size_t)B_ * H_ * N_ * R_);
  _Float16* v16  = carve((size_t)B_ * H_ * N_ * DK_);
  _Float16* z16  = carve(BN * D_);

  auto cvt = [&](const float* src, _Float16* dst, int n) {
    cvt_f32_to_f16<<<(n + 255) / 256, 256, 0, stream>>>(src, dst, n);
  };
  cvt(x,  x16,  (int)(BN * D_));
  cvt(Wq, Wq16, D_ * D_);
  cvt(Wk, Wk16, D_ * D_);
  cvt(Wv, Wv16, D_ * D_);
  cvt(Wp, Wp16, D_ * D_);
  cvt(U,  U16,  DK_ * R_);

  qkv_kernel<<<dim3((unsigned)(BN / 128), H_, 3), 256, 0, stream>>>(
      x16, Wq16, Wk16, Wv16, U16, q16, k16, v16);
  attn_kernel<<<dim3(N_ / 64, H_, B_), 128, 0, stream>>>(q16, k16, v16, z16);
  proj_kernel<<<dim3((unsigned)(BN / 128), D_ / 64), 256, 0, stream>>>(
      z16, Wp16, out);
}
